// MultiHeadAttention_66340064854505
// MI455X (gfx1250) — matrix operand order repaired and checked by reading, not compiled, not run on hardware
//
#include <hip/hip_runtime.h>

typedef __attribute__((ext_vector_type(16))) __bf16 v16bf;
typedef __attribute__((ext_vector_type(8)))  float  v8f;

#define N_SEQ 4096
#define E_DIM 512
#define H_NUM 8
#define DK    64
#define WSIZE 128
// v stored transposed: qkvb[VBASE + (h*64+d)*4096 + n]
#define VBASE (2 * H_NUM * N_SEQ * DK)

// ---------- bf16 helpers ----------
__device__ __forceinline__ unsigned short f32_to_bf16_bits(float f) {
    union { float f; unsigned int u; } v; v.f = f;
    unsigned int u = v.u;
    unsigned int r = u + 0x7FFFu + ((u >> 16) & 1u);
    return (unsigned short)(r >> 16);
}
__device__ __forceinline__ float bf16_lo_f32(unsigned int u) {   // low bf16 of a dword
    union { unsigned int u; float f; } v; v.u = u << 16; return v.f;
}
__device__ __forceinline__ float bf16_hi_f32(unsigned int u) {   // high bf16 of a dword
    union { unsigned int u; float f; } v; v.u = u & 0xFFFF0000u; return v.f;
}
__device__ __forceinline__ v8f zero8() {
    v8f z;
#pragma unroll
    for (int r = 0; r < 8; ++r) z[r] = 0.0f;
    return z;
}
__device__ __forceinline__ v8f wmma_bf16(v16bf a, v16bf b, v8f c) {
    return __builtin_amdgcn_wmma_f32_16x16x32_bf16(false, a, false, b, (short)0, c, false, false);
}

// ---------- vectorized WMMA fragment loaders (ISA 7.12.2 layouts) ----------
// A 16x32: lane->M=row0+l15; elements = two contiguous 8-element K-chunks at k0+8*half and +16.
__device__ __forceinline__ v16bf load_a_frag(const unsigned short* s, int row0, int k0, int ld, int lane) {
    const unsigned short* p = s + (row0 + (lane & 15)) * ld + k0 + 8 * (lane >> 4);
    union { uint4 u[2]; v16bf v; } t;
    t.u[0] = *(const uint4*)p;
    t.u[1] = *(const uint4*)(p + 16);
    return t.v;
}
// B 32x16 from K-contiguous (N-major) storage S[n][k]: lane->N=n0+l15; elements = two contiguous 8-element K-chunks at k0+8*half and +16.
__device__ __forceinline__ v16bf load_b_frag_t(const unsigned short* s, int n0, int k0, int ld, int lane) {
    const unsigned short* p = s + (n0 + (lane & 15)) * ld + k0 + 8 * (lane >> 4);
    union { uint4 u[2]; v16bf v; } t;
    t.u[0] = *(const uint4*)p;
    t.u[1] = *(const uint4*)(p + 16);
    return t.v;
}

// ---------- Kernel 0: convert x -> bf16; transpose+convert W_qkv, W_out ----------
__global__ void convert_bf16_kernel(const float* __restrict__ x, const float* __restrict__ wqkv,
                                    const float* __restrict__ wout,
                                    unsigned short* __restrict__ xb, unsigned short* __restrict__ wqkvT,
                                    unsigned short* __restrict__ woutT) {
    int i = blockIdx.x * blockDim.x + threadIdx.x;
    const int nx = N_SEQ * E_DIM;
    const int nq = 3 * E_DIM * E_DIM;      // W_qkv^T: [1536][512]
    const int no = E_DIM * E_DIM;          // W_out^T: [512][512]
    if (i < nx) xb[i] = f32_to_bf16_bits(x[i]);
    if (i < nq) {
        int n = i >> 9, k = i & 511;
        wqkvT[i] = f32_to_bf16_bits(wqkv[k * (3 * E_DIM) + n]);
    }
    if (i < no) {
        int n = i >> 9, k = i & 511;
        woutT[i] = f32_to_bf16_bits(wout[k * E_DIM + n]);
    }
}

// ---------- shared GEMM machinery: 128x64 tile, BK=32, double-buffered LDS ----------
#define LDA_S 40   // 80B row stride: conflict-free 16-lane row reads
#define LDB_S 40

__device__ __forceinline__ void stage_tile(unsigned short* AsB, unsigned short* BsB,
                                           const unsigned short* __restrict__ aG,
                                           const unsigned short* __restrict__ bG,
                                           int m0, int n0, int kt, int tid) {
#pragma unroll
    for (int p = 0; p < 2; ++p) {                    // A: 128x32
        int idx = p * 256 + tid;
        int row = idx >> 2, c8 = (idx & 3) * 8;
        *(uint4*)&AsB[row * LDA_S + c8] = *(const uint4*)&aG[(m0 + row) * E_DIM + kt + c8];
    }
    {                                                // B^T: 64 rows (N) x 32 (K)
        int row = tid >> 2, c8 = (tid & 3) * 8;
        *(uint4*)&BsB[row * LDB_S + c8] = *(const uint4*)&bG[(n0 + row) * E_DIM + kt + c8];
    }
}

// ---------- Kernel 1: QKV GEMM, scatter q/k head-major, v transposed d-major ----------
__global__ __launch_bounds__(256)
void qkv_gemm_kernel(const unsigned short* __restrict__ xb, const unsigned short* __restrict__ wbT,
                     const float* __restrict__ bias, unsigned short* __restrict__ qkvb) {
    __shared__ __align__(16) unsigned short As[2][128 * LDA_S];
    __shared__ __align__(16) unsigned short Bs[2][64 * LDB_S];
    const int tid = threadIdx.x, lane = tid & 31, wave = tid >> 5;
    const int l15 = lane & 15, half = lane >> 4;
    const int wm = wave >> 1, wn = wave & 1;
    const int m0 = blockIdx.x * 128;
    const int n0 = blockIdx.y * 64;
    v8f acc[2][2];
#pragma unroll
    for (int i = 0; i < 2; ++i)
#pragma unroll
        for (int j = 0; j < 2; ++j) acc[i][j] = zero8();

    stage_tile(As[0], Bs[0], xb, wbT, m0, n0, 0, tid);
    int buf = 0;
    for (int kt = 0; kt < E_DIM; kt += 32, buf ^= 1) {
        __syncthreads();
        if (kt + 32 < E_DIM) {
            stage_tile(As[buf ^ 1], Bs[buf ^ 1], xb, wbT, m0, n0, kt + 32, tid);
            if (kt + 64 < E_DIM) {
                __builtin_prefetch(&xb[(m0 + (tid >> 2)) * E_DIM + kt + 64 + (tid & 3) * 8], 0, 1);
                __builtin_prefetch(&wbT[(n0 + (tid >> 2)) * E_DIM + kt + 64 + (tid & 3) * 8], 0, 1);
            }
        }
        v16bf bf[2];
#pragma unroll
        for (int j = 0; j < 2; ++j) bf[j] = load_b_frag_t(Bs[buf], wn * 32 + j * 16, 0, LDB_S, lane);
#pragma unroll
        for (int i = 0; i < 2; ++i) {
            v16bf a = load_a_frag(As[buf], wm * 32 + i * 16, 0, LDA_S, lane);
#pragma unroll
            for (int j = 0; j < 2; ++j) acc[i][j] = wmma_bf16(a, bf[j], acc[i][j]);
        }
    }
#pragma unroll
    for (int i = 0; i < 2; ++i)
#pragma unroll
        for (int j = 0; j < 2; ++j) {
            int col = n0 + wn * 32 + j * 16 + l15;
            float bv = bias[col];
            int which = col / E_DIM, e = col % E_DIM;
            int h = e >> 6, d = e & 63;
            int row0 = m0 + wm * 32 + i * 16 + 8 * half;
            if (which == 2) {
                // v: [h][d][n] layout, 8 consecutive rows -> one packed 16B store
                unsigned int b0 = f32_to_bf16_bits(acc[i][j][0] + bv) | ((unsigned)f32_to_bf16_bits(acc[i][j][1] + bv) << 16);
                unsigned int b1 = f32_to_bf16_bits(acc[i][j][2] + bv) | ((unsigned)f32_to_bf16_bits(acc[i][j][3] + bv) << 16);
                unsigned int b2 = f32_to_bf16_bits(acc[i][j][4] + bv) | ((unsigned)f32_to_bf16_bits(acc[i][j][5] + bv) << 16);
                unsigned int b3 = f32_to_bf16_bits(acc[i][j][6] + bv) | ((unsigned)f32_to_bf16_bits(acc[i][j][7] + bv) << 16);
                uint4 pk; pk.x = b0; pk.y = b1; pk.z = b2; pk.w = b3;
                *(uint4*)&qkvb[(size_t)VBASE + ((size_t)(h * DK + d)) * N_SEQ + row0] = pk;
            } else {
#pragma unroll
                for (int r = 0; r < 8; ++r) {
                    float v = acc[i][j][r] + bv;
                    qkvb[(((size_t)(which * H_NUM + h) * N_SEQ + row0 + r) << 6) + d] = f32_to_bf16_bits(v);
                }
            }
        }
}

// ---------- Kernel 2: per-head column sum of V (v is [h][d][n] -> contiguous) ----------
__global__ void vtot_kernel(const unsigned short* __restrict__ qkvb, float* __restrict__ vtot) {
    int i = blockIdx.x * blockDim.x + threadIdx.x;
    if (i >= H_NUM * DK) return;
    const unsigned short* vg = qkvb + (size_t)VBASE + (size_t)i * N_SEQ;
    float s = 0.0f;
    for (int n = 0; n < N_SEQ; n += 8) {
        uint4 u = *(const uint4*)&vg[n];
        s += bf16_lo_f32(u.x) + bf16_hi_f32(u.x) + bf16_lo_f32(u.y) + bf16_hi_f32(u.y) +
             bf16_lo_f32(u.z) + bf16_hi_f32(u.z) + bf16_lo_f32(u.w) + bf16_hi_f32(u.w);
    }
    vtot[i] = s;
}

// ---------- Kernel 3: CEMA scan ----------
__global__ void cema_kernel(const float* __restrict__ x, const float* __restrict__ pc,
                            const float* __restrict__ qc, const float* __restrict__ gc,
                            float* __restrict__ ys) {
    int e = blockIdx.x * blockDim.x + threadIdx.x;
    if (e >= E_DIM) return;
    float p[16], q[16], g[16], h[16];
#pragma unroll
    for (int d = 0; d < 16; ++d) {
        p[d] = pc[e * 16 + d]; q[d] = qc[e * 16 + d]; g[d] = gc[e * 16 + d]; h[d] = 0.0f;
    }
    for (int t = 0; t < N_SEQ; ++t) {
        float xt = x[t * E_DIM + e];
        float y = 0.0f;
#pragma unroll
        for (int d = 0; d < 16; ++d) {
            h[d] = fmaf(q[d], h[d], p[d] * xt);
            y = fmaf(g[d], h[d], y);
        }
        ys[t * E_DIM + e] = y;
    }
}

// ---------- Kernel 4: block-local attention, multiplicative-mask softmax ----------
#define LDQK 72    // Q/K LDS row stride (144B, conflict-free)
#define LDP  136   // P  LDS row stride (272B, conflict-free)
#define LDV  136   // Vt LDS row stride
__global__ __launch_bounds__(256)
void attn_kernel(const unsigned short* __restrict__ qkvb, const float* __restrict__ vtot,
                 unsigned short* __restrict__ attnb) {
    __shared__ __align__(16) unsigned short smem[2 * 128 * LDQK];  // Q|K, later aliased by P (128*LDP fits)
    __shared__ __align__(16) unsigned short Vt[64 * LDV];          // V transposed: [d][j]
    unsigned short* Qs = smem;
    unsigned short* Ks = smem + 128 * LDQK;
    unsigned short* Ps = smem;

    const int h = blockIdx.x >> 5;
    const int blk = blockIdx.x & 31;
    const int tid = threadIdx.x, lane = tid & 31, wave = tid >> 5;
    const int l15 = lane & 15, half = lane >> 4;
    const unsigned short* qg  = qkvb + (((size_t)(0 * H_NUM + h) * N_SEQ + blk * WSIZE) << 6);
    const unsigned short* kg  = qkvb + (((size_t)(1 * H_NUM + h) * N_SEQ + blk * WSIZE) << 6);
    const unsigned short* vg0 = qkvb + (size_t)VBASE + (size_t)h * DK * N_SEQ + blk * WSIZE;
#pragma unroll
    for (int p = 0; p < 4; ++p) {                      // Q,K: 128x64 row-major tiles
        int idx = p * 256 + tid;
        int row = idx >> 3, c8 = (idx & 7) * 8;
        *(uint4*)&Qs[row * LDQK + c8] = *(const uint4*)&qg[row * 64 + c8];
        *(uint4*)&Ks[row * LDQK + c8] = *(const uint4*)&kg[row * 64 + c8];
    }
#pragma unroll
    for (int p = 0; p < 4; ++p) {                      // Vt: 64x128 (d-major, contiguous in j)
        int idx = p * 256 + tid;
        int row = idx >> 4, c8 = (idx & 15) * 8;
        *(uint4*)&Vt[row * LDV + c8] = *(const uint4*)&vg0[(size_t)row * N_SEQ + c8];
    }
    __syncthreads();

    const int mBase = wave * 16;
    // ---- S = (Q K^T) * dk^-1/2 ----
    v8f st[8];
    {
        v16bf aq0 = load_a_frag(Qs, mBase, 0, LDQK, lane);
        v16bf aq1 = load_a_frag(Qs, mBase, 32, LDQK, lane);
#pragma unroll
        for (int t = 0; t < 8; ++t) {
            v8f c = zero8();
            c = wmma_bf16(aq0, load_b_frag_t(Ks, t * 16, 0, LDQK, lane), c);
            c = wmma_bf16(aq1, load_b_frag_t(Ks, t * 16, 32, LDQK, lane), c);
#pragma unroll
            for (int r = 0; r < 8; ++r) st[t][r] = c[r] * 0.125f;
        }
    }
    __syncthreads();   // everyone done with Q/K before P aliases them

    // ---- multiplicative-mask softmax (zeros outside window contribute exp(0)) ----
    float p0v[8];
#pragma unroll
    for (int r = 0; r < 8; ++r) {
        const int row = mBase + r + 8 * half;
        float m = 0.0f;
#pragma unroll
        for (int t = 0; t < 8; ++t) {
            int col = t * 16 + l15;
            if (col <= row) m = fmaxf(m, st[t][r]);
        }
#pragma unroll
        for (int off = 1; off < 16; off <<= 1) m = fmaxf(m, __shfl_xor(m, off, 32));
        float zs = 0.0f;
#pragma unroll
        for (int t = 0; t < 8; ++t) {
            int col = t * 16 + l15;
            if (col <= row) zs += __expf(st[t][r] - m);
        }
#pragma unroll
        for (int off = 1; off < 16; off <<= 1) zs += __shfl_xor(zs, off, 32);
        float e0 = __expf(-m);
        float z = zs + (float)(N_SEQ - row - 1) * e0;
        float p0 = e0 / z;
        p0v[r] = p0;
#pragma unroll
        for (int t = 0; t < 8; ++t) {
            int col = t * 16 + l15;
            float pv = (col <= row) ? (__expf(st[t][r] - m) / z - p0) : 0.0f;
            Ps[row * LDP + col] = f32_to_bf16_bits(pv);
        }
    }
    __syncthreads();

    // ---- O = P' V + p0 * Vtot ----
    v8f oacc[4];
#pragma unroll
    for (int t = 0; t < 4; ++t) oacc[t] = zero8();
#pragma unroll
    for (int kc = 0; kc < 128; kc += 32) {
        v16bf ap = load_a_frag(Ps, mBase, kc, LDP, lane);
#pragma unroll
        for (int t = 0; t < 4; ++t)
            oacc[t] = wmma_bf16(ap, load_b_frag_t(Vt, t * 16, kc, LDV, lane), oacc[t]);
    }
#pragma unroll
    for (int t = 0; t < 4; ++t) {
        int d = t * 16 + l15;
        float vt = vtot[h * DK + d];
#pragma unroll
        for (int r = 0; r < 8; ++r) {
            int grow = blk * WSIZE + mBase + r + 8 * half;
            float val = oacc[t][r] + p0v[r] * vt;
            attnb[(size_t)grow * E_DIM + h * DK + d] = f32_to_bf16_bits(val);
        }
    }
}

// ---------- Kernel 5: out-projection GEMM + adaptive mix with CEMA ----------
__global__ __launch_bounds__(256)
void out_gemm_kernel(const unsigned short* __restrict__ ab, const unsigned short* __restrict__ wbT,
                     const float* __restrict__ bias, const float* __restrict__ cema,
                     const float* __restrict__ aw, float* __restrict__ out) {
    __shared__ __align__(16) unsigned short As[2][128 * LDA_S];
    __shared__ __align__(16) unsigned short Bs[2][64 * LDB_S];
    const int tid = threadIdx.x, lane = tid & 31, wave = tid >> 5;
    const int l15 = lane & 15, half = lane >> 4;
    const int wm = wave >> 1, wn = wave & 1;
    const int m0 = blockIdx.x * 128;
    const int n0 = blockIdx.y * 64;
    v8f acc[2][2];
#pragma unroll
    for (int i = 0; i < 2; ++i)
#pragma unroll
        for (int j = 0; j < 2; ++j) acc[i][j] = zero8();

    stage_tile(As[0], Bs[0], ab, wbT, m0, n0, 0, tid);
    int buf = 0;
    for (int kt = 0; kt < E_DIM; kt += 32, buf ^= 1) {
        __syncthreads();
        if (kt + 32 < E_DIM) {
            stage_tile(As[buf ^ 1], Bs[buf ^ 1], ab, wbT, m0, n0, kt + 32, tid);
            if (kt + 64 < E_DIM) {
                __builtin_prefetch(&ab[(m0 + (tid >> 2)) * E_DIM + kt + 64 + (tid & 3) * 8], 0, 1);
                __builtin_prefetch(&wbT[(n0 + (tid >> 2)) * E_DIM + kt + 64 + (tid & 3) * 8], 0, 1);
            }
        }
        v16bf bf[2];
#pragma unroll
        for (int j = 0; j < 2; ++j) bf[j] = load_b_frag_t(Bs[buf], wn * 32 + j * 16, 0, LDB_S, lane);
#pragma unroll
        for (int i = 0; i < 2; ++i) {
            v16bf a = load_a_frag(As[buf], wm * 32 + i * 16, 0, LDA_S, lane);
#pragma unroll
            for (int j = 0; j < 2; ++j) acc[i][j] = wmma_bf16(a, bf[j], acc[i][j]);
        }
    }
    float a0 = aw[0], a1 = aw[1];
    float mx = fmaxf(a0, a1);
    float e0 = __expf(a0 - mx), e1 = __expf(a1 - mx);
    float w0 = e0 / (e0 + e1), w1 = e1 / (e0 + e1);
#pragma unroll
    for (int i = 0; i < 2; ++i)
#pragma unroll
        for (int j = 0; j < 2; ++j) {
            int col = n0 + wn * 32 + j * 16 + l15;
            float bv = bias[col];
#pragma unroll
            for (int r = 0; r < 8; ++r) {
                int row = m0 + wm * 32 + i * 16 + r + 8 * half;
                float v = acc[i][j][r] + bv;
                out[(size_t)row * E_DIM + col] = w0 * v + w1 * cema[(size_t)row * E_DIM + col];
            }
        }
}

// ---------- host-side launcher ----------
extern "C" void kernel_launch(void* const* d_in, const int* in_sizes, int n_in,
                              void* d_out, int out_size, void* d_ws, size_t ws_size,
                              hipStream_t stream) {
    (void)in_sizes; (void)n_in; (void)out_size; (void)ws_size;
    const float* x    = (const float*)d_in[0];
    const float* Wqkv = (const float*)d_in[1];
    const float* bqkv = (const float*)d_in[2];
    const float* Wout = (const float*)d_in[3];
    const float* bout = (const float*)d_in[4];
    /* d_in[5] = omega : unused by the reference math */
    const float* pc   = (const float*)d_in[6];
    const float* qc   = (const float*)d_in[7];
    const float* gc   = (const float*)d_in[8];
    const float* aw   = (const float*)d_in[9];
    float* out = (float*)d_out;
    char* ws = (char*)d_ws;

    // workspace layout (bytes)
    unsigned short* xb    = (unsigned short*)(ws + 0);          //  4,194,304  x bf16
    unsigned short* wqkvT = (unsigned short*)(ws + 4194304);    //  1,572,864  W_qkv^T bf16 [1536][512]
    unsigned short* woutT = (unsigned short*)(ws + 5767168);    //    524,288  W_out^T bf16 [512][512]
    unsigned short* qkvb  = (unsigned short*)(ws + 6291456);    // 12,582,912  q/k head-major, v d-major bf16
    unsigned short* attnb = (unsigned short*)(ws + 18874368);   //  4,194,304  attn out bf16 [4096][512]
    float*          vtot  = (float*)        (ws + 23068672);    //      2,048  per-head V col sums
    float*          cema  = (float*)        (ws + 23070720);    //  8,388,608  CEMA output f32

    convert_bf16_kernel<<<8192, 256, 0, stream>>>(x, Wqkv, Wout, xb, wqkvT, woutT);

    dim3 g1(N_SEQ / 128, (3 * E_DIM) / 64);   // 32 x 24
    qkv_gemm_kernel<<<g1, 256, 0, stream>>>(xb, wqkvT, bqkv, qkvb);

    vtot_kernel<<<2, 256, 0, stream>>>(qkvb, vtot);
    cema_kernel<<<2, 256, 0, stream>>>(x, pc, qc, gc, cema);

    attn_kernel<<<H_NUM * (N_SEQ / WSIZE), 256, 0, stream>>>(qkvb, vtot, attnb);

    dim3 g2(N_SEQ / 128, E_DIM / 64);         // 32 x 8
    out_gemm_kernel<<<g2, 256, 0, stream>>>(attnb, woutT, bout, cema, aw, out);
}
